// GraphSeqGenerator_77799037599897
// MI455X (gfx1250) — matrix operand order repaired and checked by reading, not compiled, not run on hardware
//
#include <hip/hip_runtime.h>
#include <functional>

typedef __attribute__((ext_vector_type(16))) _Float16 v16h;
typedef __attribute__((ext_vector_type(8)))  float    v8f;

// ---------------- problem constants ----------------
#define NN    2048
#define EE    65536
#define TOBS  6
#define FIN   2
#define HENC  256
#define LATD  128
#define HDEC  256
#define PREDN 6
#define VISR  0.5f
#define EWSC  2.0f      /* 1/(MAX_EW-MIN_EW) */
#define XPAD  64        /* padded feature width for x / y_hat */
#define YPADC 16        /* padded output cols for dec_lin */
#define KC    64        /* K-chunk staged per LDS round */
#define BSTR  68        /* LDS stride (floats) for transposed B tile: bank-optimal */

// Async global->LDS staging when the gfx1250 builtins are available.
#if defined(__gfx1250__) && __has_builtin(__builtin_amdgcn_global_load_async_to_lds_b32) && __has_builtin(__builtin_amdgcn_s_wait_asynccnt)
#define USE_ASYNC_LDS 1
#else
#define USE_ASYNC_LDS 0
#endif

typedef __attribute__((address_space(1))) int gint_t;
typedef __attribute__((address_space(3))) int lint_t;

__device__ __forceinline__ void stage_one(const float* g, float* l) {
#if USE_ASYNC_LDS
  __builtin_amdgcn_global_load_async_to_lds_b32((gint_t*)g, (lint_t*)l, 0, 0);
#else
  *l = *g;
#endif
}
__device__ __forceinline__ void stage_join() {
#if USE_ASYNC_LDS
  __builtin_amdgcn_s_wait_asynccnt(0);
#endif
}

__device__ __forceinline__ void put4(v16h& v, int o, float4 q, bool relu) {
  if (relu) {
    q.x = fmaxf(q.x, 0.0f); q.y = fmaxf(q.y, 0.0f);
    q.z = fmaxf(q.z, 0.0f); q.w = fmaxf(q.w, 0.0f);
  }
  v[o + 0] = (_Float16)q.x; v[o + 1] = (_Float16)q.y;
  v[o + 2] = (_Float16)q.z; v[o + 3] = (_Float16)q.w;
}

// ---------------------------------------------------------------------------
// C[NN,Ncols] = act( A0@B0 + A1@B1 + A2@B2 + A3@B3 + bias0 + bias1 )
// Row-major. Requirements (guaranteed by host-side padding):
//   Nrows == NN (2048), K % 64 == 0, Ncols % 16 == 0, lda == K, ldb == Ncols.
// Block = 256 threads = 8 waves -> 128x16 output slab; B tile shared via LDS.
// act: 0 none, 1 sigmoid, 2 tanh.  reluA: ReLU applied to A0 elements.
// ---------------------------------------------------------------------------
__global__ void __launch_bounds__(256) wmma_gemm(
    const float* __restrict__ A0, const float* __restrict__ B0, int K0,
    const float* __restrict__ A1, const float* __restrict__ B1, int K1,
    const float* __restrict__ A2, const float* __restrict__ B2, int K2,
    const float* __restrict__ A3, const float* __restrict__ B3, int K3,
    const float* __restrict__ bias0, const float* __restrict__ bias1,
    float* __restrict__ C, int Ncols, int act, int reluA) {
  __shared__ float lf[16 * BSTR];   // transposed B tile: lf[c*BSTR + k], k<64
  const int tid  = threadIdx.x;
  const int wave = tid >> 5;
  const int lane = tid & 31;
  const int hi   = lane >> 4;
  const int ln   = lane & 15;
  const int colTiles = Ncols >> 4;
  const int bTc = blockIdx.x % colTiles;
  const int bTr = blockIdx.x / colTiles;
  const int tr  = bTr * 8 + wave;        // this wave's 16-row tile
  const int rowA = tr * 16 + ln;
  const int colB = bTc * 16;

  v8f acc = {};
  const float* As[4] = {A0, A1, A2, A3};
  const float* Bs[4] = {B0, B1, B2, B3};
  const int    Ks[4] = {K0, K1, K2, K3};
#pragma unroll
  for (int p = 0; p < 4; ++p) {
    const float* A = As[p];
    if (A == nullptr) continue;           // uniform branch
    const float* B = Bs[p];
    const int K = Ks[p];
    const bool ra = (p == 0) && reluA;
    const float* arow = A + (size_t)rowA * (size_t)K;
    for (int kb = 0; kb < K; kb += KC) {
      __builtin_prefetch(arow + kb + KC, 0, 1);  // speculative A prefetch
      __syncthreads();                            // protect prior LDS reads
      // stage B[kb..kb+63, colB..colB+15] -> lf[c*BSTR + k] (coalesced loads)
#pragma unroll
      for (int i = 0; i < 4; ++i) {
        int e = tid + i * 256;                    // 1024 elements
        int c = e & 15, k = e >> 4;
        stage_one(B + (size_t)(kb + k) * (size_t)Ncols + (colB + c),
                  &lf[c * BSTR + k]);
      }
      stage_join();
      __syncthreads();
#pragma unroll
      for (int kk = 0; kk < 2; ++kk) {
        // ---- A fragment (ISA 16-bit A 16x32 layout) ----
        const float* ap = arow + kb + kk * 32 + hi * 8;
        v16h a;
        put4(a, 0,  *(const float4*)(ap + 0),  ra);
        put4(a, 4,  *(const float4*)(ap + 4),  ra);
        put4(a, 8,  *(const float4*)(ap + 16), ra);
        put4(a, 12, *(const float4*)(ap + 20), ra);
        // ---- B fragment from LDS (ISA 16-bit B 32x16 layout, matches A) ----
        const float4* bp = (const float4*)&lf[ln * BSTR + kk * 32 + hi * 8];
        v16h b;
        put4(b, 0,  bp[0], false);
        put4(b, 4,  bp[1], false);
        put4(b, 8,  bp[4], false);
        put4(b, 12, bp[5], false);
        acc = __builtin_amdgcn_wmma_f32_16x16x32_f16(false, a, false, b,
                                                     (short)0, acc, false, false);
      }
    }
  }

  const int col = colB + ln;
  float bias = 0.0f;
  if (bias0) bias += bias0[col];
  if (bias1) bias += bias1[col];
#pragma unroll
  for (int j = 0; j < 8; ++j) {
    int r = tr * 16 + hi * 8 + j;        // D layout: VGPR j -> M = hi*8 + j
    float v = acc[j] + bias;
    if (act == 1)      v = 1.0f / (1.0f + expf(-v));
    else if (act == 2) v = tanhf(v);
    C[(size_t)r * (size_t)Ncols + col] = v;
  }
}

// ---------------- utility / sparse-graph kernels ----------------
__global__ void k_fill0(float* p, int n) {
  int i = blockIdx.x * blockDim.x + threadIdx.x;
  if (i < n) p[i] = 0.0f;
}
__global__ void k_copy(float* dst, const float* src, int n) {
  int i = blockIdx.x * blockDim.x + threadIdx.x;
  if (i < n) dst[i] = src[i];
}
// [n,2] -> padded [n,XPAD] (cols 0,1; rest must be pre-zeroed)
__global__ void k_pack2(const float* s, float* d, int n) {
  int i = blockIdx.x * blockDim.x + threadIdx.x;
  if (i < n * 2) { int r = i >> 1, c = i & 1; d[r * XPAD + c] = s[i]; }
}
// padded [n,YPADC] cols 0,1 -> [n,2]
__global__ void k_unpack2(const float* s, float* d, int n) {
  int i = blockIdx.x * blockDim.x + threadIdx.x;
  if (i < n * 2) { int r = i >> 1, c = i & 1; d[i] = s[r * YPADC + c]; }
}
// enc_Wx [6 slices][FIN,HENC] -> [6][XPAD,HENC] (pad rows pre-zeroed)
__global__ void k_pad_wx(const float* s, float* d) {
  int i = blockIdx.x * blockDim.x + threadIdx.x;
  if (i < 6 * FIN * HENC) {
    int sl = i / (FIN * HENC), rem = i - sl * (FIN * HENC);
    int r = rem / HENC, c = rem - r * HENC;
    d[(size_t)sl * XPAD * HENC + r * HENC + c] = s[i];
  }
}
// dec_lin_W [HDEC,2] -> [HDEC,YPADC] (pad cols pre-zeroed)
__global__ void k_pad_wy(const float* s, float* d) {
  int i = blockIdx.x * blockDim.x + threadIdx.x;
  if (i < HDEC * 2) { int r = i >> 1, c = i & 1; d[r * YPADC + c] = s[i]; }
}
__global__ void k_deg_sparse(const int* src, const float* w, float* deg, int E) {
  int e = blockIdx.x * blockDim.x + threadIdx.x;
  if (e < E) atomicAdd(&deg[src[e]], w[e]);
}
__global__ void k_dinv(const float* deg, float* dinv, int n) {
  int i = blockIdx.x * blockDim.x + threadIdx.x;
  if (i < n) { float d = deg[i]; dinv[i] = (d > 0.0f) ? rsqrtf(d) : 0.0f; }
}
__global__ void k_edge_norm(const int* src, const int* dst, const float* w,
                            const float* dinv, float* nrm, int E) {
  int e = blockIdx.x * blockDim.x + threadIdx.x;
  if (e < E) nrm[e] = -dinv[src[e]] * w[e] * dinv[dst[e]];
}
// out[dst,f] += nrm * X[src,f], f < F, row stride ld
__global__ void k_spmm(const int* src, const int* dst, const float* nrm,
                       const float* X, float* out, int E, int F, int ld) {
  int idx = blockIdx.x * blockDim.x + threadIdx.x;
  if (idx < E * F) {
    int e = idx / F, f = idx - e * F;
    float v = nrm[e] * X[(size_t)src[e] * ld + f];
    atomicAdd(&out[(size_t)dst[e] * ld + f], v);
  }
}

// ---------------- dense (radius) graph kernels ----------------
__global__ void k_dense_w_deg(const float* yhat, float* M, float* deg, int n) {
  __shared__ float red[256];
  int i = blockIdx.x;
  float xi = yhat[i * 2 + 0], yi = yhat[i * 2 + 1];  // MIN=0,MAX=1 -> identity
  float sum = 0.0f;
  for (int j = threadIdx.x; j < n; j += blockDim.x) {
    float dx = xi - yhat[j * 2 + 0];
    float dy = yi - yhat[j * 2 + 1];
    float d = sqrtf(dx * dx + dy * dy);
    float w = (d < VISR && d > 0.0f) ? d * EWSC : 0.0f;
    M[(size_t)i * (size_t)n + j] = w;
    sum += w;
  }
  red[threadIdx.x] = sum;
  __syncthreads();
  for (int s = blockDim.x >> 1; s > 0; s >>= 1) {
    if (threadIdx.x < (unsigned)s) red[threadIdx.x] += red[threadIdx.x + s];
    __syncthreads();
  }
  if (threadIdx.x == 0) deg[i] = red[0];
}
__global__ void k_dense_scale(const float* dinv, float* M, int n) {
  int idx = blockIdx.x * blockDim.x + threadIdx.x;
  if (idx < n * n) {
    int i = idx / n, j = idx - i * n;
    M[idx] = -(dinv[i] * M[idx]) * dinv[j];
  }
}

// ---------------- GRU elementwise ----------------
__global__ void k_mul(const float* a, const float* b, float* c, int n) {
  int i = blockIdx.x * blockDim.x + threadIdx.x;
  if (i < n) c[i] = a[i] * b[i];
}
__global__ void k_gru_combine(const float* z, const float* ht, float* H, int n) {
  int i = blockIdx.x * blockDim.x + threadIdx.x;
  if (i < n) { float zz = z[i]; H[i] = zz * H[i] + (1.0f - zz) * ht[i]; }
}

// ---------------- host orchestration ----------------
extern "C" void kernel_launch(void* const* d_in, const int* in_sizes, int n_in,
                              void* d_out, int out_size, void* d_ws, size_t ws_size,
                              hipStream_t stream) {
  const float* x_seq  = (const float*)d_in[0];
  const int*   ei_seq = (const int*)  d_in[1];
  const float* ew_seq = (const float*)d_in[2];
  const float* h_enc0 = (const float*)d_in[3];
  const float* h_dec0 = (const float*)d_in[4];
  const float* enc_Wx = (const float*)d_in[5];
  const float* enc_bx = (const float*)d_in[6];
  const float* enc_Wh = (const float*)d_in[7];
  const float* enc_bh = (const float*)d_in[8];
  const float* enc_lW = (const float*)d_in[9];
  const float* enc_lb = (const float*)d_in[10];
  const float* dec_Wx = (const float*)d_in[11];
  const float* dec_bx = (const float*)d_in[12];
  const float* dec_Wh = (const float*)d_in[13];
  const float* dec_bh = (const float*)d_in[14];
  const float* dec_lW = (const float*)d_in[15];
  const float* dec_lb = (const float*)d_in[16];
  float* out = (float*)d_out;

  // workspace layout (floats)
  float* w = (float*)d_ws;
  float* M    = w; w += (size_t)NN * NN;
  float* txa  = w; w += (size_t)NN * HENC;   // propagated x (stride = ldx)
  float* txh  = w; w += (size_t)NN * HENC;
  float* txhr = w; w += (size_t)NN * HENC;
  float* Gz   = w; w += (size_t)NN * HENC;
  float* Gr   = w; w += (size_t)NN * HENC;
  float* Gh   = w; w += (size_t)NN * HENC;
  float* Hr   = w; w += (size_t)NN * HENC;
  float* he   = w; w += (size_t)NN * HENC;
  float* hd   = w; w += (size_t)NN * HDEC;
  float* zlat = w; w += (size_t)NN * LATD;
  float* xpad = w; w += (size_t)NN * XPAD;
  float* ypad = w; w += (size_t)NN * YPADC;
  float* Wxp  = w; w += (size_t)6 * XPAD * HENC;
  float* lWyp = w; w += (size_t)HDEC * YPADC;
  float* deg  = w; w += NN;
  float* dinv = w; w += NN;
  float* nrm  = w; w += EE;

  auto fill = [&](float* p, int n) {
    k_fill0<<<(n + 255) / 256, 256, 0, stream>>>(p, n);
  };
  auto gemm = [&](const float* A0, const float* B0, int K0,
                  const float* A1, const float* B1, int K1,
                  const float* A2, const float* B2, int K2,
                  const float* A3, const float* B3, int K3,
                  const float* b0, const float* b1,
                  float* C, int Nc, int act, int reluA) {
    int blocks = (NN / 128) * (Nc / 16);   // 8 waves/block, 128x16 slab
    wmma_gemm<<<blocks, 256, 0, stream>>>(A0, B0, K0, A1, B1, K1, A2, B2, K2,
                                          A3, B3, K3, b0, b1, C, Nc, act, reluA);
  };

  // prop(X, Fdata, ld, O): O[NN,ld] = P @ X  (P = negated normalized adjacency)
  using PropFn = std::function<void(const float*, int, int, float*)>;

  auto gru = [&](const float* x, int Fd, int ldx, float* H, int Hdim,
                 const float* Wx, const float* bx, const float* Wh, const float* bh,
                 const PropFn& prop) {
    size_t wxs = (size_t)ldx * Hdim;
    size_t whs = (size_t)Hdim * Hdim;
    prop(x, Fd, ldx, txa);
    prop(H, Hdim, Hdim, txh);
    gemm(x, Wx + 0 * wxs, ldx, txa, Wx + 1 * wxs, ldx,
         H, Wh + 0 * whs, Hdim, txh, Wh + 1 * whs, Hdim,
         bx + 0 * Hdim, bh + 0 * Hdim, Gz, Hdim, 1, 0);
    gemm(x, Wx + 2 * wxs, ldx, txa, Wx + 3 * wxs, ldx,
         H, Wh + 2 * whs, Hdim, txh, Wh + 3 * whs, Hdim,
         bx + 1 * Hdim, bh + 1 * Hdim, Gr, Hdim, 1, 0);
    int n = NN * Hdim;
    k_mul<<<(n + 255) / 256, 256, 0, stream>>>(H, Gr, Hr, n);
    prop(Hr, Hdim, Hdim, txhr);
    gemm(x, Wx + 4 * wxs, ldx, txa, Wx + 5 * wxs, ldx,
         Hr, Wh + 4 * whs, Hdim, txhr, Wh + 5 * whs, Hdim,
         bx + 2 * Hdim, bh + 2 * Hdim, Gh, Hdim, 2, 0);
    k_gru_combine<<<(n + 255) / 256, 256, 0, stream>>>(Gz, Gh, H, n);
  };

  auto step = [&](const PropFn& prop, float* ydst) {
    gru(xpad, FIN, XPAD, he, HENC, Wxp, enc_bx, enc_Wh, enc_bh, prop);
    gemm(he, enc_lW, HENC, nullptr, nullptr, 0, nullptr, nullptr, 0,
         nullptr, nullptr, 0, enc_lb, nullptr, zlat, LATD, 0, 1);
    gru(zlat, LATD, LATD, hd, HDEC, dec_Wx, dec_bx, dec_Wh, dec_bh, prop);
    if (ydst) {
      gemm(hd, lWyp, HDEC, nullptr, nullptr, 0, nullptr, nullptr, 0,
           nullptr, nullptr, 0, dec_lb, nullptr, ypad, YPADC, 0, 1);
      k_unpack2<<<(NN * 2 + 255) / 256, 256, 0, stream>>>(ypad, ydst, NN);
    }
  };

  // per-launch init (no reliance on leftover workspace state)
  fill(xpad, NN * XPAD);
  fill(Wxp, 6 * XPAD * HENC);
  fill(lWyp, HDEC * YPADC);
  k_pad_wx<<<(6 * FIN * HENC + 255) / 256, 256, 0, stream>>>(enc_Wx, Wxp);
  k_pad_wy<<<(HDEC * 2 + 255) / 256, 256, 0, stream>>>(dec_lW, lWyp);
  k_copy<<<(NN * HENC + 255) / 256, 256, 0, stream>>>(he, h_enc0, NN * HENC);
  k_copy<<<(NN * HDEC + 255) / 256, 256, 0, stream>>>(hd, h_dec0, NN * HDEC);

  // ---- observed phase: sparse edge-list graphs ----
  for (int t = 0; t < TOBS; ++t) {
    const int* srcp = ei_seq + (size_t)t * 2 * EE;
    const int* dstp = srcp + EE;
    const float* wp = ew_seq + (size_t)t * EE;
    fill(deg, NN);
    k_deg_sparse<<<EE / 256, 256, 0, stream>>>(srcp, wp, deg, EE);
    k_dinv<<<(NN + 255) / 256, 256, 0, stream>>>(deg, dinv, NN);
    k_edge_norm<<<EE / 256, 256, 0, stream>>>(srcp, dstp, wp, dinv, nrm, EE);
    PropFn prop = [&, srcp, dstp](const float* X, int Fd, int ld, float* O) {
      fill(O, NN * ld);
      int tot = EE * Fd;
      k_spmm<<<(tot + 255) / 256, 256, 0, stream>>>(srcp, dstp, nrm, X, O,
                                                    EE, Fd, ld);
    };
    k_pack2<<<(NN * 2 + 255) / 256, 256, 0, stream>>>(
        x_seq + (size_t)t * NN * FIN, xpad, NN);
    step(prop, (t == TOBS - 1) ? out : nullptr);
  }

  // ---- rollout phase: dense radius graph from predictions ----
  for (int p = 1; p < PREDN; ++p) {
    const float* yprev = out + (size_t)(p - 1) * NN * FIN;
    k_dense_w_deg<<<NN, 256, 0, stream>>>(yprev, M, deg, NN);
    k_dinv<<<(NN + 255) / 256, 256, 0, stream>>>(deg, dinv, NN);
    k_dense_scale<<<(NN * NN + 255) / 256, 256, 0, stream>>>(dinv, M, NN);
    PropFn prop = [&](const float* X, int Fd, int ld, float* O) {
      gemm(M, X, NN, nullptr, nullptr, 0, nullptr, nullptr, 0,
           nullptr, nullptr, 0, nullptr, nullptr, O, ld, 0, 0);
    };
    k_pack2<<<(NN * 2 + 255) / 256, 256, 0, stream>>>(yprev, xpad, NN);
    step(prop, out + (size_t)p * NN * FIN);
  }
}
